// Head_63359357550851
// MI455X (gfx1250) — matrix operand order repaired and checked by reading, not compiled, not run on hardware
//
#include <hip/hip_runtime.h>

// Problem constants (from reference): B=4, T=4096, C=1024, H=64
#define BB 4
#define TT 4096
#define CC 1024
#define HH 64
#define BT (BB * TT)

typedef __attribute__((ext_vector_type(16))) __bf16 v16bf;
typedef __attribute__((ext_vector_type(8)))  __bf16 v8bf;
typedef __attribute__((ext_vector_type(8)))  float  v8f;
typedef __attribute__((ext_vector_type(4)))  float  v4f;

// fp32 -> bf16 via native convert (backend emits v_cvt_pk_bf16_f32)
static __device__ __forceinline__ unsigned short f2bfu(float f) {
  return __builtin_bit_cast(unsigned short, (__bf16)f);
}

// ---------------------------------------------------------------------------
// Kernel 0: repack W (fp32 [C,H]) into pre-swizzled bf16 WMMA B-fragments.
// Fragment id f = (sel*4 + ntile)*32 + kstep ; blob = 32 lanes x 16 bf16.
// B-frag layout (16x16x32 bf16): b[j] = B[(j&7) + hi*8 + (j>>3)*16][lane&15]
// ---------------------------------------------------------------------------
__global__ void __launch_bounds__(32) wfrag_build(
    const float* __restrict__ Wq, const float* __restrict__ Wk,
    const float* __restrict__ Wv, unsigned short* __restrict__ wfrag) {
  int f = blockIdx.x;             // 0..383
  int sel = f >> 7;               // 0..2
  int ntile = (f >> 5) & 3;       // 0..3
  int kstep = f & 31;             // 0..31
  const float* W = (sel == 0) ? Wq : ((sel == 1) ? Wk : Wv);
  int lane = threadIdx.x & 31;
  int lm = lane & 15, hi = lane >> 4;
  unsigned short* dst = wfrag + (size_t)f * 512 + lane * 16;
  int col = ntile * 16 + lm;
#pragma unroll
  for (int j = 0; j < 16; ++j) {
    int row = kstep * 32 + (j & 7) + hi * 8 + (j >> 3) * 16;     // K index within C
    dst[j] = f2bfu(W[row * HH + col]);
  }
}

// ---------------------------------------------------------------------------
// Kernel 1: QKV projection. One wave per 16-row tile of x; x read ONCE from
// HBM, fed to 12 accumulators (Q/K/V x 4 column tiles of 16).
// B-fragments are software-pipelined with two ping-pong register buffers:
// the k-loop is unrolled by 2 giving 6 stages/body; stage g computes with
// one buffer while the loads for stage g+1 fill the other. 6 stages is even
// so buffer parity is loop-invariant -> no rotation copies, and each load
// clause is covered by 4 WMMAs instead of stalling before every WMMA.
// Q pre-scaled by C^-0.5 = 2^-5 (exact in bf16). V stored transposed [B,H,T].
// ---------------------------------------------------------------------------
__global__ void __launch_bounds__(128) qkv_proj(
    const float* __restrict__ x, const unsigned short* __restrict__ wfrag,
    unsigned short* __restrict__ qb, unsigned short* __restrict__ kb,
    unsigned short* __restrict__ vt) {
  int wave = threadIdx.x >> 5;
  int rt = blockIdx.x * 4 + wave;       // row tile 0..1023
  int m0 = rt * 16;                     // global token base
  int lane = threadIdx.x & 31;
  int lm = lane & 15, hi = lane >> 4;

  v8f acc[12];
  v8f zero = {0.f, 0.f, 0.f, 0.f, 0.f, 0.f, 0.f, 0.f};
#pragma unroll
  for (int m = 0; m < 12; ++m) acc[m] = zero;

  // A-frag: a[j] = x[m0+lm][kk + (j&7) + hi*8 + (j>>3)*16]
  const float* xrow = x + (size_t)(m0 + lm) * CC + hi * 8;
  const unsigned short* wl = wfrag + lane * 16;

  // ping-pong fragment buffers; prologue loads stage 0 (grp 0, kstep 0)
  v16bf b0[4], b1[4];
#pragma unroll
  for (int m = 0; m < 4; ++m)
    b0[m] = *(const v16bf*)(wl + (size_t)(m * 32) * 512);

  for (int ks2 = 0; ks2 < 16; ++ks2) {    // two k-steps (of 32) per body
    // ---- load & convert two A-fragments (x read once from HBM) ----
    const float* xp = xrow + ks2 * 64;
    v4f e0 = *(const v4f*)(xp);
    v4f e1 = *(const v4f*)(xp + 4);
    v4f e2 = *(const v4f*)(xp + 16);
    v4f e3 = *(const v4f*)(xp + 20);
    v4f o0 = *(const v4f*)(xp + 32);
    v4f o1 = *(const v4f*)(xp + 36);
    v4f o2 = *(const v4f*)(xp + 48);
    v4f o3 = *(const v4f*)(xp + 52);
    v16bf a0, a1;
#pragma unroll
    for (int j = 0; j < 4; ++j) {
      a0[j] = (__bf16)e0[j]; a0[j + 4]  = (__bf16)e1[j];
      a0[j + 8] = (__bf16)e2[j]; a0[j + 12] = (__bf16)e3[j];
      a1[j] = (__bf16)o0[j]; a1[j + 4]  = (__bf16)o1[j];
      a1[j + 8] = (__bf16)o2[j]; a1[j + 12] = (__bf16)o3[j];
    }

    // ---- 6 pipelined stages: grp = g%3, kstep = 2*ks2 + (g>=3) ----
#pragma unroll
    for (int g = 0; g < 6; ++g) {
      // issue loads for the NEXT stage (stage 0 of next body when g==5).
      // At ks2==15,g==5 this reads fragment idx m*32+32 <= 128 < 384:
      // harmless in-bounds read of wfrag, never consumed.
      int ngrp = (g + 1) % 3;
      int nks = 2 * ks2 + ((g + 1) / 3);
      v16bf* nb = ((g + 1) & 1) ? b1 : b0;
      const v16bf* cb = (g & 1) ? b1 : b0;
#pragma unroll
      for (int m = 0; m < 4; ++m)
        nb[m] = *(const v16bf*)(wl + (size_t)((ngrp * 4 + m) * 32 + nks) * 512);
      const v16bf& av = (g < 3) ? a0 : a1;
#pragma unroll
      for (int m = 0; m < 4; ++m) {
        int ai = (g % 3) * 4 + m;
        acc[ai] = __builtin_amdgcn_wmma_f32_16x16x32_bf16(
            false, av, false, cb[m], (short)0, acc[ai], false, false);
      }
    }
  }

  // C/D layout: acc[..][i] = tile[row = i + 8*hi][col = lm]
#pragma unroll
  for (int nt = 0; nt < 4; ++nt) {
    int col = nt * 16 + lm;
#pragma unroll
    for (int i = 0; i < 8; ++i) {
      int g = m0 + i + 8 * hi;                         // global token
      qb[(size_t)g * HH + col] = f2bfu(acc[nt][i] * 0.03125f);   // fold scale
      kb[(size_t)g * HH + col] = f2bfu(acc[4 + nt][i]);
      int b = g >> 12, t = g & (TT - 1);
      vt[((size_t)b * HH + col) * TT + t] = f2bfu(acc[8 + nt][i]); // V^T
    }
  }
}

// ---------------------------------------------------------------------------
// Kernel 2: flash attention. 4 waves/block, each wave owns 16 query rows.
// Per 32-key block: K and V fragments are ALL issued up-front (V loads are
// independent of S, their latency hides under the softmax VALU stage), then
// 4 WMMAs for S = Q K^T, online softmax via shfl_xor butterflies over the
// 16-lane halves, P re-swizzled C-frag -> A-frag through padded LDS,
// 4 WMMAs for O += P V. K/V stream from L2 (512 KB/batch, shared).
// ---------------------------------------------------------------------------
__global__ void __launch_bounds__(128) attn(
    const unsigned short* __restrict__ qb, const unsigned short* __restrict__ kb,
    const unsigned short* __restrict__ vt, float* __restrict__ out) {
  __shared__ unsigned short plds[4 * 16 * 40];   // per-wave 16 x (32+8 pad)
  int wave = threadIdx.x >> 5;
  int lane = threadIdx.x & 31;
  int lm = lane & 15, hi = lane >> 4;
  int bb = blockIdx.y;
  int qt0 = blockIdx.x * 64 + wave * 16;         // query row base (in batch)
  size_t g0 = (size_t)bb * TT + qt0;

  // Q A-fragments for the two h-chunks (h 0..31 and 32..63), kept in regs
  v16bf qf[2];
  const unsigned short* qrow = qb + (g0 + lm) * HH + hi * 16;
#pragma unroll
  for (int kc = 0; kc < 2; ++kc) {
    v8bf lo = *(const v8bf*)(qrow + kc * 32);
    v8bf h8 = *(const v8bf*)(qrow + kc * 32 + 8);
#pragma unroll
    for (int j = 0; j < 8; ++j) { qf[kc][j] = lo[j]; qf[kc][j + 8] = h8[j]; }
  }

  v8f zero = {0.f, 0.f, 0.f, 0.f, 0.f, 0.f, 0.f, 0.f};
  v8f o[4];
#pragma unroll
  for (int n = 0; n < 4; ++n) o[n] = zero;
  float m_s[8], l_s[8];
#pragma unroll
  for (int i = 0; i < 8; ++i) { m_s[i] = -1e30f; l_s[i] = 0.f; }

  unsigned short* pw = plds + wave * (16 * 40);
  const unsigned short* kbase = kb + (size_t)bb * TT * HH;
  const unsigned short* vbase = vt + (size_t)bb * HH * TT;

  int nkb = (qt0 + 47) >> 5;                     // causal: keys 0..qt0+15
  for (int kblk = 0; kblk < nkb; ++kblk) {
    int ks = kblk * 32;
    if (kblk + 1 < nkb) {                        // global_prefetch_b8 next K/V
      __builtin_prefetch(kbase + (size_t)(ks + 32 + lane) * HH, 0, 1);
      __builtin_prefetch(vbase + (size_t)(lane * 2) * TT + ks + 32, 0, 1);
    }

    // ---- issue ALL K and V fragment loads for this key block up front ----
    // K-frag: b[j] = K[ks + sub*16 + lm][kc*32 + j + hi*16]
    v16bf kf[4];
#pragma unroll
    for (int kc = 0; kc < 2; ++kc) {
      kf[kc * 2 + 0] = *(const v16bf*)(kbase + (size_t)(ks + lm) * HH + kc * 32 + hi * 16);
      kf[kc * 2 + 1] = *(const v16bf*)(kbase + (size_t)(ks + 16 + lm) * HH + kc * 32 + hi * 16);
    }
    // V-frag: b[j] = V[ks + j + hi*16][n*16 + lm] via V^T rows
    v16bf vf[4];
#pragma unroll
    for (int n = 0; n < 4; ++n)
      vf[n] = *(const v16bf*)(vbase + (size_t)(n * 16 + lm) * TT + ks + hi * 16);

    // ---- S = Q K^T (scale already folded into Q) ----
    v8f sc0 = zero, sc1 = zero;
#pragma unroll
    for (int kc = 0; kc < 2; ++kc) {
      sc0 = __builtin_amdgcn_wmma_f32_16x16x32_bf16(
          false, qf[kc], false, kf[kc * 2 + 0], (short)0, sc0, false, false);
      sc1 = __builtin_amdgcn_wmma_f32_16x16x32_bf16(
          false, qf[kc], false, kf[kc * 2 + 1], (short)0, sc1, false, false);
    }

    // ---- online softmax (VALU stage hides the V-load latency) ----
    bool need_mask = (ks + 31) > qt0;            // uniform per wave
    float p0[8], p1[8];
#pragma unroll
    for (int i = 0; i < 8; ++i) {
      float v0 = sc0[i], v1 = sc1[i];
      if (need_mask) {
        int t = qt0 + i + 8 * hi;
        if (ks + lm > t)      v0 = -1e30f;
        if (ks + 16 + lm > t) v1 = -1e30f;
      }
      float mx = fmaxf(v0, v1);                  // row max over 16 lanes
#pragma unroll
      for (int d = 1; d < 16; d <<= 1) mx = fmaxf(mx, __shfl_xor(mx, d, 32));
      float mnew = fmaxf(m_s[i], mx);
      float alpha = __expf(m_s[i] - mnew);
      float e0 = __expf(v0 - mnew);
      float e1 = __expf(v1 - mnew);
      float rs = e0 + e1;                        // row sum over 16 lanes
#pragma unroll
      for (int d = 1; d < 16; d <<= 1) rs += __shfl_xor(rs, d, 32);
      l_s[i] = l_s[i] * alpha + rs;
      m_s[i] = mnew;
      p0[i] = e0; p1[i] = e1;
#pragma unroll
      for (int n = 0; n < 4; ++n) o[n][i] *= alpha;
    }

    // P: C-frag -> LDS [row][col], then reload in A-frag order (same wave,
    // DS ops complete in order; padded stride 40 keeps 16B alignment).
#pragma unroll
    for (int i = 0; i < 8; ++i) {
      int r = (i + 8 * hi) * 40;
      pw[r + lm]      = f2bfu(p0[i]);
      pw[r + 16 + lm] = f2bfu(p1[i]);
    }
    v16bf pf;
    {
      const unsigned short* prow = pw + lm * 40 + hi * 16;
      v8bf lo = *(const v8bf*)(prow);
      v8bf h8 = *(const v8bf*)(prow + 8);
#pragma unroll
      for (int j = 0; j < 8; ++j) { pf[j] = lo[j]; pf[j + 8] = h8[j]; }
    }

    // ---- O += P V (V fragments already resident) ----
#pragma unroll
    for (int n = 0; n < 4; ++n)
      o[n] = __builtin_amdgcn_wmma_f32_16x16x32_bf16(
          false, pf, false, vf[n], (short)0, o[n], false, false);
  }

  // normalize + store fp32 [B,T,H]
#pragma unroll
  for (int i = 0; i < 8; ++i) {
    float inv = 1.0f / l_s[i];
    size_t g = g0 + i + 8 * hi;
#pragma unroll
    for (int n = 0; n < 4; ++n)
      out[g * HH + n * 16 + lm] = o[n][i] * inv;
  }
}

// ---------------------------------------------------------------------------
extern "C" void kernel_launch(void* const* d_in, const int* in_sizes, int n_in,
                              void* d_out, int out_size, void* d_ws, size_t ws_size,
                              hipStream_t stream) {
  (void)in_sizes; (void)n_in; (void)out_size; (void)ws_size;
  const float* x  = (const float*)d_in[0];
  const float* Wq = (const float*)d_in[1];
  const float* Wk = (const float*)d_in[2];
  const float* Wv = (const float*)d_in[3];
  float* out = (float*)d_out;

  // workspace: [0, 384KB) W fragments ; then Q, K (bf16 [B,T,H]) ; V^T ([B,H,T])
  char* ws = (char*)d_ws;
  unsigned short* wfrag = (unsigned short*)ws;
  unsigned short* qb = (unsigned short*)(ws + 384 * 1024);
  unsigned short* kb = qb + (size_t)BT * HH;
  unsigned short* vt = kb + (size_t)BT * HH;

  wfrag_build<<<dim3(384), dim3(32), 0, stream>>>(Wq, Wk, Wv, wfrag);
  qkv_proj<<<dim3(BT / 64), dim3(128), 0, stream>>>(x, wfrag, qb, kb, vt);
  attn<<<dim3(TT / 64, BB), dim3(128), 0, stream>>>(qb, kb, vt, out);
}
